// ConformerMHSARelPosV2_76836964925500
// MI455X (gfx1250) — matrix operand order repaired and checked by reading, not compiled, not run on hardware
//
#include <hip/hip_runtime.h>
#include <hip/hip_bf16.h>
#include <math.h>

typedef __attribute__((ext_vector_type(16))) _Float16 v16h;
typedef __attribute__((ext_vector_type(8)))  float    v8f;

#define B_  8
#define T_  1024
#define E_  512
#define H_  8
#define D_  64
#define L_PAD 2048          // padded 2T-1 = 2047 -> 2048 rows
#define LN_EPS 1e-5f

// ---------- WMMA helpers (CDNA5 gfx1250, wave32) ----------

static __device__ __forceinline__ v8f wmma32(v16h a, v16h b, v8f c) {
  // v_wmma_f32_16x16x32_f16 : D = A(16x32 f16) * B(32x16 f16) + C(16x16 f32)
  return __builtin_amdgcn_wmma_f32_16x16x32_f16(false, a, false, b, (short)0, c, false, false);
}

// A-matrix 16x32 f16 fragment, contiguous k order (matches load_b_frag):
// lanes 0-15 (M=lane) hold K=0..15; lanes 16-31 hold K=16..31.
// => per lane: halves 0..15 = K(k0 + hi*16 + 0..15).
static __device__ __forceinline__ v16h load_a_frag(const _Float16* base, int ld, int k0) {
  int lane = threadIdx.x & 31;
  int m  = lane & 15;
  int hi = lane >> 4;
  const _Float16* r = base + (size_t)m * ld + k0 + hi * 16;
  union { uint4 u[2]; v16h v; } u;
  u.u[0] = *reinterpret_cast<const uint4*>(r);
  u.u[1] = *reinterpret_cast<const uint4*>(r + 8);
  return u.v;
}

// B-matrix 32x16 f16 fragment, B[k][n]. Contiguous k order: lanes 0-15 hold K=0..15 of
// column n=lane; lanes 16-31 hold K=16..31 of column n=lane-16.
// base points at element [n=0][k=0] of a (N x K) row-major "transposed" array.
static __device__ __forceinline__ v16h load_b_frag(const _Float16* base, int ld, int k0) {
  int lane = threadIdx.x & 31;
  int n  = lane & 15;
  int hi = lane >> 4;
  const _Float16* r = base + (size_t)n * ld + k0 + hi * 16;
  union { uint4 u[2]; v16h v; } u;
  u.u[0] = *reinterpret_cast<const uint4*>(r);
  u.u[1] = *reinterpret_cast<const uint4*>(r + 8);
  return u.v;
}

// ---------- LayerNorm: fp32 in -> f16 out ----------
__global__ void k_layernorm(const float* __restrict__ x, const float* __restrict__ g,
                            const float* __restrict__ be, _Float16* __restrict__ xh) {
  int row = blockIdx.x;
  int tid = threadIdx.x;              // 256 threads, 2 elems each (E=512)
  const float* xr = x + (size_t)row * E_;
  float a0 = xr[tid], a1 = xr[tid + 256];
  __shared__ float red[256];
  red[tid] = a0 + a1;
  __syncthreads();
  for (int s = 128; s > 0; s >>= 1) { if (tid < s) red[tid] += red[tid + s]; __syncthreads(); }
  float mu = red[0] * (1.0f / E_);
  __syncthreads();
  float d0 = a0 - mu, d1 = a1 - mu;
  red[tid] = d0 * d0 + d1 * d1;
  __syncthreads();
  for (int s = 128; s > 0; s >>= 1) { if (tid < s) red[tid] += red[tid + s]; __syncthreads(); }
  float rstd = rsqrtf(red[0] * (1.0f / E_) + LN_EPS);
  _Float16* o = xh + (size_t)row * E_;
  o[tid]       = (_Float16)(d0 * rstd * g[tid]       + be[tid]);
  o[tid + 256] = (_Float16)(d1 * rstd * g[tid + 256] + be[tid + 256]);
}

// ---------- Sinusoidal PE rows (pos = T-1-p), f16 ----------
__global__ void k_pe(_Float16* __restrict__ pe) {
  int p = blockIdx.x;                 // 0..L_PAD-1 (row 2047 is padding, never read)
  int m = threadIdx.x;                // 0..255
  float pos  = (float)(T_ - 1 - p);
  float freq = powf(10000.0f, -(2.0f * (float)m) / (float)E_);
  float ph   = pos * freq;
  pe[(size_t)p * E_ + 2 * m]     = (_Float16)sinf(ph);
  pe[(size_t)p * E_ + 2 * m + 1] = (_Float16)cosf(ph);
}

// ---------- Weight transpose + convert: (K x N) f32 -> (N x K) f16 ----------
__global__ void k_transpose_cvt(const float* __restrict__ w, _Float16* __restrict__ wt,
                                int K, int N) {
  int idx = blockIdx.x * blockDim.x + threadIdx.x;
  if (idx >= K * N) return;
  int k = idx / N, n = idx - k * N;
  wt[(size_t)n * K + k] = (_Float16)w[idx];
}

// ---------- Generic WMMA GEMM: C(MxN) = A(MxK,f16) @ Wt^T (Wt is NxK,f16) ----------
// block = 128 threads (4 waves) covering a 64(M) x 64(N) tile; each wave owns a
// 16x64 strip: 4 accumulators, A fragment reused across 4 B fragments per k-step
// (2x the FLOP/byte of a 16x16-per-wave scheme; QKV GEMM is the largest FLOP block).
template<bool F32OUT>
__global__ void __launch_bounds__(128) k_gemm(const _Float16* __restrict__ A,
                                              const _Float16* __restrict__ Wt,
                                              void* __restrict__ out,
                                              int M, int N, int K) {
  int wave = threadIdx.x >> 5;
  int lane = threadIdx.x & 31;
  int m0 = blockIdx.y * 64 + wave * 16;
  int n0 = blockIdx.x * 64;
  v8f acc0 = {}, acc1 = {}, acc2 = {}, acc3 = {};
  const _Float16* abase = A  + (size_t)m0 * K;
  const _Float16* b0 = Wt + (size_t)(n0)      * K;
  const _Float16* b1 = Wt + (size_t)(n0 + 16) * K;
  const _Float16* b2 = Wt + (size_t)(n0 + 32) * K;
  const _Float16* b3 = Wt + (size_t)(n0 + 48) * K;
  for (int k0 = 0; k0 < K; k0 += 32) {
    v16h af  = load_a_frag(abase, K, k0);
    v16h bf0 = load_b_frag(b0, K, k0);
    v16h bf1 = load_b_frag(b1, K, k0);
    v16h bf2 = load_b_frag(b2, K, k0);
    v16h bf3 = load_b_frag(b3, K, k0);
    acc0 = wmma32(af, bf0, acc0);
    acc1 = wmma32(af, bf1, acc1);
    acc2 = wmma32(af, bf2, acc2);
    acc3 = wmma32(af, bf3, acc3);
  }
  int n = lane & 15, hi = lane >> 4;
#pragma unroll
  for (int r = 0; r < 8; ++r) {
    size_t o = (size_t)(m0 + r + hi * 8) * N + n0 + n;
    if (F32OUT) {
      ((float*)out)[o]      = acc0[r];
      ((float*)out)[o + 16] = acc1[r];
      ((float*)out)[o + 32] = acc2[r];
      ((float*)out)[o + 48] = acc3[r];
    } else {
      ((_Float16*)out)[o]      = (_Float16)acc0[r];
      ((_Float16*)out)[o + 16] = (_Float16)acc1[r];
      ((_Float16*)out)[o + 32] = (_Float16)acc2[r];
      ((_Float16*)out)[o + 48] = (_Float16)acc3[r];
    }
  }
}

// ---------- q+u, q+v (f16) and V transpose to (B,H,D,T) ----------
__global__ void k_prep_qv(const _Float16* __restrict__ qkv,
                          const float* __restrict__ ub, const float* __restrict__ vb,
                          _Float16* __restrict__ qu, _Float16* __restrict__ qv,
                          _Float16* __restrict__ vT) {
  int idx = blockIdx.x * blockDim.x + threadIdx.x;
  if (idx >= B_ * T_ * E_) return;
  int e  = idx % E_;
  int bt = idx / E_;
  int b  = bt / T_, t = bt - b * T_;
  int h  = e / D_,  f = e - h * D_;
  float q = (float)qkv[(size_t)bt * (3 * E_) + e];
  qu[idx] = (_Float16)(q + ub[h * D_ + f]);
  qv[idx] = (_Float16)(q + vb[h * D_ + f]);
  vT[(((size_t)b * H_ + h) * D_ + f) * T_ + t] = qkv[(size_t)bt * (3 * E_) + 2 * E_ + e];
}

// ---------- Fused flash attention with Transformer-XL rel-shift ----------
// One wave per (b, h, 16-query tile). Online softmax over key blocks of 32.
// Scores: S = (q+u)K^T + shift((q+v)R^T); shift(i,j) -> rel index T-1-i+j.
__global__ void __launch_bounds__(32) k_attn(const _Float16* __restrict__ qu,
                                             const _Float16* __restrict__ qv,
                                             const _Float16* __restrict__ qkv,
                                             const _Float16* __restrict__ rel,
                                             const _Float16* __restrict__ vT,
                                             const unsigned char* __restrict__ mask,
                                             _Float16* __restrict__ outp) {
  const int NT = T_ / 16;
  int idx = blockIdx.x;
  int it = idx % NT;
  int h  = (idx / NT) % H_;
  int b  = idx / (NT * H_);
  int i0 = it * 16;
  int lane = threadIdx.x;
  int n = lane & 15, hi = lane >> 4;

  __shared__ __align__(16) float    Slds[16][32];   // AC scores for 32 keys
  __shared__ __align__(16) float    BDlds[16][48];  // wide BD band (47 used)
  __shared__ __align__(16) _Float16 Plds[16][32];   // probabilities (f16)
  __shared__ float aLds[16];                        // per-row rescale alpha
  __shared__ float ilLds[16];                       // per-row 1/l

  const _Float16* qub = qu + (size_t)(b * T_ + i0) * E_ + h * D_;
  const _Float16* qvb = qv + (size_t)(b * T_ + i0) * E_ + h * D_;
  v16h qu0 = load_a_frag(qub, E_, 0);
  v16h qu1 = load_a_frag(qub, E_, 32);
  v16h qv0 = load_a_frag(qvb, E_, 0);
  v16h qv1 = load_a_frag(qvb, E_, 32);

  v8f o0 = {}, o1 = {}, o2 = {}, o3 = {};
  float mreg = -3.0e38f, lreg = 0.0f;   // row stats (valid on lanes 0..15)

  const _Float16* kbase = qkv + (size_t)(b * T_) * (3 * E_) + E_ + h * D_;  // K rows
  const _Float16* vtb   = vT + ((size_t)b * H_ + h) * D_ * T_;              // V^T rows
  const unsigned char* mrow = mask + (size_t)b * T_;

  for (int j0 = 0; j0 < T_; j0 += 32) {
    // ---- AC = (q+u) K^T : two 16x16 tiles ----
#pragma unroll
    for (int t2 = 0; t2 < 2; ++t2) {
      const _Float16* kb = kbase + (size_t)(j0 + t2 * 16) * (3 * E_);
      v8f ac = {};
      ac = wmma32(qu0, load_b_frag(kb, 3 * E_, 0),  ac);
      ac = wmma32(qu1, load_b_frag(kb, 3 * E_, 32), ac);
#pragma unroll
      for (int r = 0; r < 8; ++r) Slds[r + hi * 8][t2 * 16 + n] = ac[r];
    }
    // ---- BDwide = (q+v) R^T over 48-wide rel band ----
    int base = (T_ - 16) - i0 + j0;       // rel row for wide-col 0 (>= 0 always)
#pragma unroll
    for (int t3 = 0; t3 < 3; ++t3) {
      int rrow = base + t3 * 16 + n;      // this lane's wide column -> rel row
      if (rrow > 2 * T_ - 2) rrow = 2 * T_ - 2;   // col 47 pad, never read back
      const _Float16* rb = rel + (size_t)rrow * E_ + h * D_;
      union { uint4 u[2]; v16h v; } b0, b1;
      const _Float16* rp0 = rb + hi * 16;
      b0.u[0] = *reinterpret_cast<const uint4*>(rp0);
      b0.u[1] = *reinterpret_cast<const uint4*>(rp0 + 8);
      const _Float16* rp1 = rb + 32 + hi * 16;
      b1.u[0] = *reinterpret_cast<const uint4*>(rp1);
      b1.u[1] = *reinterpret_cast<const uint4*>(rp1 + 8);
      v8f bd = {};
      bd = wmma32(qv0, b0.v, bd);
      bd = wmma32(qv1, b1.v, bd);
#pragma unroll
      for (int r = 0; r < 8; ++r) BDlds[r + hi * 8][t3 * 16 + n] = bd[r];
    }
    __syncthreads();

    // ---- online softmax (lanes 0..15, one row each) ----
    if (lane < 16) {
      int row = lane;
      float sv[32];
      float vmax = mreg;
#pragma unroll
      for (int j = 0; j < 32; ++j) {
        float s = Slds[row][j] + BDlds[row][j + 15 - row];   // rel-shift gather
        s += mrow[j0 + j] ? 0.0f : -1.0e30f;                 // sequence mask
        s *= 0.125f;                                         // 1/sqrt(D)
        sv[j] = s;
        vmax = fmaxf(vmax, s);
      }
      float alpha = expf(mreg - vmax);
      float psum = 0.0f;
#pragma unroll
      for (int j = 0; j < 32; ++j) {
        float pr = expf(sv[j] - vmax);
        psum += pr;
        Plds[row][j] = (_Float16)pr;
      }
      lreg = lreg * alpha + psum;
      mreg = vmax;
      aLds[row] = alpha;
    }
    __syncthreads();

    // ---- rescale O, then O += P @ V (4 WMMAs: D=64) ----
    v16h pf = load_a_frag(&Plds[0][0], 32, 0);
#pragma unroll
    for (int r = 0; r < 8; ++r) {
      float a = aLds[r + hi * 8];
      o0[r] *= a; o1[r] *= a; o2[r] *= a; o3[r] *= a;
    }
    o0 = wmma32(pf, load_b_frag(vtb + (size_t)0  * T_, T_, j0), o0);
    o1 = wmma32(pf, load_b_frag(vtb + (size_t)16 * T_, T_, j0), o1);
    o2 = wmma32(pf, load_b_frag(vtb + (size_t)32 * T_, T_, j0), o2);
    o3 = wmma32(pf, load_b_frag(vtb + (size_t)48 * T_, T_, j0), o3);
    __syncthreads();
  }

  if (lane < 16) ilLds[lane] = 1.0f / lreg;
  __syncthreads();
  _Float16* ob = outp + (size_t)(b * T_ + i0) * E_ + h * D_;
#pragma unroll
  for (int r = 0; r < 8; ++r) {
    int rr = r + hi * 8;
    float inv = ilLds[rr];
    ob[(size_t)rr * E_ +  0 + n] = (_Float16)(o0[r] * inv);
    ob[(size_t)rr * E_ + 16 + n] = (_Float16)(o1[r] * inv);
    ob[(size_t)rr * E_ + 32 + n] = (_Float16)(o2[r] * inv);
    ob[(size_t)rr * E_ + 48 + n] = (_Float16)(o3[r] * inv);
  }
}

// ---------- host-side orchestration ----------
extern "C" void kernel_launch(void* const* d_in, const int* in_sizes, int n_in,
                              void* d_out, int out_size, void* d_ws, size_t ws_size,
                              hipStream_t stream) {
  (void)in_sizes; (void)n_in; (void)out_size; (void)ws_size;
  const float*         x      = (const float*)d_in[0];
  const unsigned char* mask   = (const unsigned char*)d_in[1];   // bool (B,T)
  const float*         gam    = (const float*)d_in[2];
  const float*         bet    = (const float*)d_in[3];
  const float*         qkv_w  = (const float*)d_in[4];
  const float*         lpos_w = (const float*)d_in[5];
  const float*         ubias  = (const float*)d_in[6];
  const float*         vbias  = (const float*)d_in[7];
  const float*         out_w  = (const float*)d_in[8];

  char* p = (char*)d_ws;
  auto alloc = [&](size_t elems) {
    _Float16* r = (_Float16*)p;
    p += ((elems * 2 + 255) / 256) * 256;
    return r;
  };
  _Float16* xh    = alloc((size_t)B_ * T_ * E_);      // normalized x, f16
  _Float16* qkvT  = alloc((size_t)3 * E_ * E_);       // qkv_w^T f16 (3E x E)
  _Float16* lposT = alloc((size_t)E_ * E_);           // linear_pos_w^T f16
  _Float16* outT  = alloc((size_t)E_ * E_);           // out_w^T f16
  _Float16* pe    = alloc((size_t)L_PAD * E_);        // sinusoidal PE f16
  _Float16* qkv   = alloc((size_t)B_ * T_ * 3 * E_);  // q|k|v f16
  _Float16* relb  = alloc((size_t)L_PAD * E_);        // rel = pe @ lpos_w f16
  _Float16* quB   = alloc((size_t)B_ * T_ * E_);      // q + u
  _Float16* qvB   = alloc((size_t)B_ * T_ * E_);      // q + v
  _Float16* vTb   = alloc((size_t)B_ * T_ * E_);      // V^T (B,H,D,T)
  _Float16* aout  = alloc((size_t)B_ * T_ * E_);      // attention output f16

  k_layernorm<<<B_ * T_, 256, 0, stream>>>(x, gam, bet, xh);
  k_pe<<<L_PAD, 256, 0, stream>>>(pe);
  int nw;
  nw = E_ * 3 * E_; k_transpose_cvt<<<(nw + 255) / 256, 256, 0, stream>>>(qkv_w, qkvT, E_, 3 * E_);
  nw = E_ * E_;     k_transpose_cvt<<<(nw + 255) / 256, 256, 0, stream>>>(lpos_w, lposT, E_, E_);
                    k_transpose_cvt<<<(nw + 255) / 256, 256, 0, stream>>>(out_w,  outT,  E_, E_);

  // qkv = xh @ qkv_w   (8192 x 1536, K=512): grid (N/64, M/64)
  k_gemm<false><<<dim3((3 * E_) / 64, (B_ * T_) / 64), 128, 0, stream>>>(
      xh, qkvT, qkv, B_ * T_, 3 * E_, E_);
  // rel = pe @ linear_pos_w  (2048 x 512, K=512)
  k_gemm<false><<<dim3(E_ / 64, L_PAD / 64), 128, 0, stream>>>(
      pe, lposT, relb, L_PAD, E_, E_);

  nw = B_ * T_ * E_;
  k_prep_qv<<<(nw + 255) / 256, 256, 0, stream>>>(qkv, ubias, vbias, quB, qvB, vTb);

  k_attn<<<B_ * H_ * (T_ / 16), 32, 0, stream>>>(quB, qvB, qkv, relb, vTb, mask, aout);

  // final projection -> fp32 d_out  (8192 x 512, K=512)
  k_gemm<true><<<dim3(E_ / 64, (B_ * T_) / 64), 128, 0, stream>>>(
      aout, outT, d_out, B_ * T_, E_, E_);
}
